// GAT_Graph_Classifier_12515534701299
// MI455X (gfx1250) — compile-verified, matrix operand order checked by reading, not run on hardware
//
#include <hip/hip_runtime.h>

typedef __attribute__((ext_vector_type(16))) _Float16 v16h;
typedef __attribute__((ext_vector_type(8)))  float    v8f;
typedef __attribute__((ext_vector_type(4)))  unsigned u32x4;
typedef __attribute__((ext_vector_type(8)))  int      i32x8;
typedef __attribute__((ext_vector_type(4)))  int      i32x4;

#define NNODES 50000
#define NEDGES 800000
#define NGRAPH 128
#define FDIM   128     // H*D
#define NHEAD  4
#define HDIM   32
#define NCLS   10
#define NEG_SLOPE 0.2f

// ---------- helpers ----------
__device__ __forceinline__ unsigned enc_f32(float f) {
    unsigned b = __float_as_uint(f);
    return (b & 0x80000000u) ? ~b : (b | 0x80000000u);
}
__device__ __forceinline__ float dec_f32(unsigned u) {
    unsigned b = (u & 0x80000000u) ? (u & 0x7fffffffu) : ~u;
    return __uint_as_float(b);
}
__device__ __forceinline__ float lrelu(float v) {
    return v > 0.f ? v : NEG_SLOPE * v;
}

// ---------- generic fill ----------
__global__ void fill_u32_kernel(unsigned* __restrict__ p, unsigned v, int n) {
    int i = blockIdx.x * blockDim.x + threadIdx.x;
    if (i < n) p[i] = v;
}

// ---------- convert W [K=128][N=128] f32 -> transposed f16 [N][K] ----------
__global__ void wcvt_kernel(const float* __restrict__ W, _Float16* __restrict__ Wth) {
    int i = blockIdx.x * blockDim.x + threadIdx.x;   // i = n*128 + k
    if (i >= FDIM * FDIM) return;
    int n = i >> 7, k = i & 127;
    Wth[i] = (_Float16)W[k * FDIM + n];
}

// ---------- GEMM: Z[N,128] = X[N,128] @ W[128,128], WMMA f32<-f16 ----------
// One block = 16 output rows; 8 waves each own a 16-col tile (covers 128 cols).
// Weight panel (f16, [N][K]) is staged into LDS by the Tensor Data Mover with
// hardware row padding (pitch 136 halves) for bank-conflict-free fragment loads.
__global__ __launch_bounds__(256) void gemm_kernel(const float* __restrict__ X,
                                                   const _Float16* __restrict__ Wth,
                                                   float* __restrict__ Z) {
    constexpr int AP = 136;   // LDS pitch (halves): 128 + 8 skew
    constexpr int WP = 136;
    __shared__ _Float16 Alds[16 * AP];
    __shared__ _Float16 Wlds[FDIM * WP];

    const int tid  = threadIdx.x;
    const int row0 = blockIdx.x * 16;

    // --- TDM: async-load the 128x128 f16 weight panel into LDS (wave 0 only).
    // D# per CDNA5 ISA ch.8: 2-D tensor, data_size=2B, pad every 64 DWORDs
    // (one 256B row) by 4 DWORDs (16B) -> LDS pitch 272B == 136 halves.
    if (tid < 32) {
        unsigned lds_base = (unsigned)(uintptr_t)(&Wlds[0]);   // LDS byte offset
        unsigned long long ga = (unsigned long long)(uintptr_t)Wth;
        u32x4 g0 = { 1u,                                        // count=1 (valid)
                     lds_base,                                  // lds_addr
                     (unsigned)(ga & 0xffffffffu),              // global_addr[31:0]
                     ((unsigned)((ga >> 32) & 0x01ffffffu))     // global_addr[56:32]
                         | 0x80000000u };                       // type=2 ("image")
        i32x8 g1 = { (int)((1u << 16)      // data_size = 1 -> 2 bytes
                         | (1u << 20)      // pad_enable
                         | (5u << 22)      // pad_interval: 64 DWORDs
                         | (3u << 25)),    // pad_amount:   4 DWORDs
                     (int)(128u << 16),    // tensor_dim0[15:0] in bits [63:48]
                     (int)(128u << 16),    // tensor_dim0 hi=0 | tensor_dim1 lo
                     (int)(128u << 16),    // tensor_dim1 hi=0 | tile_dim0=128
                     128,                  // tile_dim1=128 | tile_dim2=0
                     128,                  // tensor_dim0_stride[31:0] = 128
                     0, 0 };               // dim0_stride hi, dim1_stride (unused)
        i32x4 gz  = { 0, 0, 0, 0 };
        i32x8 gz8 = { 0, 0, 0, 0, 0, 0, 0, 0 };
        __builtin_amdgcn_tensor_load_to_lds(g0, g1, gz, gz, gz8, 0);
        __builtin_amdgcn_s_wait_tensorcnt(0);
    }

    // --- X strip (16 rows) -> LDS with f32->f16 conversion (all 256 threads)
    for (int idx = tid; idx < 16 * FDIM; idx += 256) {
        int r = idx >> 7, k = idx & 127;
        Alds[r * AP + k] = (_Float16)X[(row0 + r) * FDIM + k];
    }
    __syncthreads();

    const int wave = tid >> 5;
    const int lane = tid & 31;
    const bool hi  = lane >= 16;          // ISA 16-bit A layout: K +8 for lanes 16..31
    const int  m   = lane & 15;           // A row / C column index
    const int  n0  = (wave << 4) + m;     // global B column for this lane

    v8f acc = {};
    #pragma unroll
    for (int kk = 0; kk < FDIM; kk += 32) {
        v16h a, b;
        #pragma unroll
        for (int v = 0; v < 8; ++v) {
            // half-pair base K for VGPR v (ISA 7.12.2, 16-bit A 16x32 layout)
            int kb = kk + 2 * v + (v >= 4 ? 8 : 0) + (hi ? 8 : 0);
            a[2 * v]     = Alds[m * AP + kb];
            a[2 * v + 1] = Alds[m * AP + kb + 1];
            b[2 * v]     = Wlds[n0 * WP + kb];
            b[2 * v + 1] = Wlds[n0 * WP + kb + 1];
        }
        acc = __builtin_amdgcn_wmma_f32_16x16x32_f16(
                  false, a, false, b, (short)0, acc, false, false);
    }

    const int mb = row0 + (hi ? 8 : 0);   // C/D layout: VGPR r -> row r (+8 for hi lanes)
    #pragma unroll
    for (int r = 0; r < 8; ++r)
        Z[(mb + r) * FDIM + (wave << 4) + m] = acc[r];
}

// ---------- attention logits: el/er [N,H] ----------
__global__ void elr_kernel(const float* __restrict__ Z,
                           const float* __restrict__ al, const float* __restrict__ ar,
                           float* __restrict__ el, float* __restrict__ er) {
    int i = blockIdx.x * blockDim.x + threadIdx.x;     // n*H + h
    if (i >= NNODES * NHEAD) return;
    int n = i >> 2, h = i & 3;
    const float* zp  = Z + n * FDIM + h * HDIM;
    const float* alp = al + h * HDIM;
    const float* arp = ar + h * HDIM;
    float sl = 0.f, sr = 0.f;
    #pragma unroll
    for (int d = 0; d < HDIM; ++d) { float zv = zp[d]; sl += zv * alp[d]; sr += zv * arp[d]; }
    el[i] = sl;
    er[i] = sr;
}

// ---------- edge pass 1: segment max over dst ----------
__global__ void edge_max_kernel(const float* __restrict__ el, const float* __restrict__ er,
                                const int* __restrict__ src, const int* __restrict__ dst,
                                unsigned* __restrict__ emaxU) {
    int e = blockIdx.x * blockDim.x + threadIdx.x;
    if (e >= NEDGES) return;
    int s = src[e], d = dst[e];
    #pragma unroll
    for (int h = 0; h < NHEAD; ++h) {
        float v = lrelu(el[s * NHEAD + h] + er[d * NHEAD + h]);
        atomicMax(&emaxU[d * NHEAD + h], enc_f32(v));
    }
}

// ---------- edge pass 2: exp + segment sum ----------
__global__ void edge_expsum_kernel(const float* __restrict__ el, const float* __restrict__ er,
                                   const int* __restrict__ src, const int* __restrict__ dst,
                                   const unsigned* __restrict__ emaxU,
                                   float* __restrict__ ee, float* __restrict__ denom) {
    int e = blockIdx.x * blockDim.x + threadIdx.x;
    if (e >= NEDGES) return;
    int s = src[e], d = dst[e];
    #pragma unroll
    for (int h = 0; h < NHEAD; ++h) {
        float v  = lrelu(el[s * NHEAD + h] + er[d * NHEAD + h]);
        float mx = dec_f32(emaxU[d * NHEAD + h]);
        float ex = __expf(v - mx);
        ee[e * NHEAD + h] = ex;
        atomicAdd(&denom[d * NHEAD + h], ex);
    }
}

// ---------- edge pass 3: alpha-weighted gather/scatter (128 lanes per edge) ----------
__global__ void edge_aggregate_kernel(const float* __restrict__ Z,
                                      const int* __restrict__ src, const int* __restrict__ dst,
                                      const float* __restrict__ ee, const float* __restrict__ denom,
                                      float* __restrict__ agg) {
    int tid = blockIdx.x * blockDim.x + threadIdx.x;   // e*128 + f
    int e = tid >> 7, f = tid & 127, h = f >> 5;
    int s = src[e], d = dst[e];
    float a = ee[e * NHEAD + h] / denom[d * NHEAD + h];
    atomicAdd(&agg[d * FDIM + f], Z[s * FDIM + f] * a);
}

// ---------- bias + relu ----------
__global__ void bias_relu_kernel(const float* __restrict__ agg, const float* __restrict__ bias,
                                 float* __restrict__ x) {
    int i = blockIdx.x * blockDim.x + threadIdx.x;
    if (i >= NNODES * FDIM) return;
    x[i] = fmaxf(agg[i] + bias[i & 127], 0.f);
}

// ---------- graph mean pooling (sums + counts) ----------
__global__ void pool_kernel(const float* __restrict__ x, const int* __restrict__ gid,
                            float* __restrict__ sums, float* __restrict__ counts) {
    int i = blockIdx.x * blockDim.x + threadIdx.x;     // n*128 + f
    int n = i >> 7, f = i & 127;
    int g = gid[n];
    atomicAdd(&sums[g * FDIM + f], x[i]);
    if (f == 0) atomicAdd(&counts[g], 1.0f);
}

// ---------- classifier head: out[G,C] = (sums/cnt) @ Wc + bc ----------
__global__ void classify_kernel(const float* __restrict__ sums, const float* __restrict__ counts,
                                const float* __restrict__ Wc, const float* __restrict__ bc,
                                float* __restrict__ out) {
    __shared__ float hg[FDIM];
    int g = blockIdx.x, t = threadIdx.x;
    float cnt = counts[g];
    hg[t] = sums[g * FDIM + t] / fmaxf(cnt, 1.0f);
    __syncthreads();
    if (t < NCLS) {
        float acc = bc[t];
        #pragma unroll 8
        for (int k = 0; k < FDIM; ++k) acc += hg[k] * Wc[k * NCLS + t];
        out[g * NCLS + t] = acc;
    }
}

extern "C" void kernel_launch(void* const* d_in, const int* in_sizes, int n_in,
                              void* d_out, int out_size, void* d_ws, size_t ws_size,
                              hipStream_t stream) {
    (void)in_sizes; (void)n_in; (void)out_size; (void)ws_size;

    const float* h_in = (const float*)d_in[0];
    const int*   src  = (const int*)d_in[1];
    const int*   dst  = (const int*)d_in[2];
    const int*   gid  = (const int*)d_in[3];
    const float* W1   = (const float*)d_in[4];
    const float* al1  = (const float*)d_in[5];
    const float* ar1  = (const float*)d_in[6];
    const float* b1   = (const float*)d_in[7];
    const float* W2   = (const float*)d_in[8];
    const float* al2  = (const float*)d_in[9];
    const float* ar2  = (const float*)d_in[10];
    const float* b2   = (const float*)d_in[11];
    const float* Wc   = (const float*)d_in[12];
    const float* bc   = (const float*)d_in[13];
    float* out = (float*)d_out;

    // carve workspace
    char* ws = (char*)d_ws;
    size_t cur = 0;
    auto carve = [&](size_t bytes) -> void* {
        void* p = ws + cur;
        cur += (bytes + 255) & ~size_t(255);
        return p;
    };
    _Float16* Wth1  = (_Float16*)carve(FDIM * FDIM * sizeof(_Float16));
    _Float16* Wth2  = (_Float16*)carve(FDIM * FDIM * sizeof(_Float16));
    float*    z     = (float*)carve((size_t)NNODES * FDIM * 4);
    float*    x     = (float*)carve((size_t)NNODES * FDIM * 4);
    float*    agg   = (float*)carve((size_t)NNODES * FDIM * 4);
    float*    el    = (float*)carve((size_t)NNODES * NHEAD * 4);
    float*    er    = (float*)carve((size_t)NNODES * NHEAD * 4);
    unsigned* emaxU = (unsigned*)carve((size_t)NNODES * NHEAD * 4);
    float*    denom = (float*)carve((size_t)NNODES * NHEAD * 4);
    float*    ee    = (float*)carve((size_t)NEDGES * NHEAD * 4);
    float*    sums  = (float*)carve((size_t)NGRAPH * FDIM * 4);
    float*    cnts  = (float*)carve((size_t)NGRAPH * 4);

    auto fillu = [&](unsigned* p, unsigned v, int n) {
        fill_u32_kernel<<<(n + 255) / 256, 256, 0, stream>>>(p, v, n);
    };

    const unsigned ENC_NEG_INF = 0x007FFFFFu;   // enc_f32(-inf)

    // one-time (per call) weight conversion to transposed f16
    wcvt_kernel<<<(FDIM * FDIM) / 256, 256, 0, stream>>>(W1, Wth1);
    wcvt_kernel<<<(FDIM * FDIM) / 256, 256, 0, stream>>>(W2, Wth2);

    auto run_layer = [&](const float* xin, const _Float16* Wth,
                         const float* al, const float* ar, const float* bias, float* xout) {
        gemm_kernel<<<NNODES / 16, 256, 0, stream>>>(xin, Wth, z);
        elr_kernel<<<(NNODES * NHEAD + 255) / 256, 256, 0, stream>>>(z, al, ar, el, er);
        fillu(emaxU, ENC_NEG_INF, NNODES * NHEAD);
        fillu((unsigned*)denom, 0u, NNODES * NHEAD);
        fillu((unsigned*)agg, 0u, NNODES * FDIM);
        edge_max_kernel<<<NEDGES / 256, 256, 0, stream>>>(el, er, src, dst, emaxU);
        edge_expsum_kernel<<<NEDGES / 256, 256, 0, stream>>>(el, er, src, dst, emaxU, ee, denom);
        edge_aggregate_kernel<<<(NEDGES * FDIM) / 256, 256, 0, stream>>>(z, src, dst, ee, denom, agg);
        bias_relu_kernel<<<(NNODES * FDIM) / 256, 256, 0, stream>>>(agg, bias, xout);
    };

    run_layer(h_in, Wth1, al1, ar1, b1, x);   // layer 1: h -> x
    run_layer(x,    Wth2, al2, ar2, b2, x);   // layer 2: x -> x (sequential, safe)

    fillu((unsigned*)sums, 0u, NGRAPH * FDIM);
    fillu((unsigned*)cnts, 0u, NGRAPH);
    pool_kernel<<<(NNODES * FDIM) / 256, 256, 0, stream>>>(x, gid, sums, cnts);
    classify_kernel<<<NGRAPH, FDIM, 0, stream>>>(sums, cnts, Wc, bc, out);
}
